// DeepConvectionNet_72748156060172
// MI455X (gfx1250) — compile-verified, matrix operand order checked by reading, not run on hardware
//
#include <hip/hip_runtime.h>
#include <math.h>

// ---------------------------------------------------------------------------
// DeepConvectionNet on MI455X (gfx1250).
// Compute-bound MLP chain (~79 GFLOP vs ~MBs of traffic) -> bf16 WMMA
// (v_wmma_f32_16x16x32_bf16) with f32 accumulation, tanh epilogue.
// Weights are DMA'd into LDS once per block via the Tensor Data Mover
// (tensor_load_to_lds + s_wait_tensorcnt); only A tiles stream per K-step.
// ---------------------------------------------------------------------------

typedef __bf16 bf16;
typedef __attribute__((ext_vector_type(16))) __bf16 v16bf;
typedef __attribute__((ext_vector_type(8)))  float  v8f;
typedef __attribute__((ext_vector_type(4)))  unsigned int v4u;
typedef __attribute__((ext_vector_type(8)))  int v8i;
typedef __attribute__((ext_vector_type(4)))  int v4i;

#define Bsz   8
#define NYc   128
#define NXc   256
#define Nc    (NYc * NXc)      // 32768
#define NFc   16384            // faces per set per batch
#define Mrows (Bsz * NFc)      // 131072 rows per face-set
#define P_ATM 101325.0f
#define EPSf  1e-8f

// padded layer widths (multiples of 32)
#define D0 256   // 245 features
#define D1 224   // 214
#define D2 192   // 185
#define D3 160   // 156
#define D4 128   // 127
#define D5 96    // 68
#define D6 32    // 10

#define LDS_STRIDE 40              // A-tile LDS row stride (80B, bank-spread)
#define A_SMEM_BYTES (256 * LDS_STRIDE * 2)

// ---------------------------------------------------------------------------
// Weight prep: f32 (din x dout, row-major) -> bf16 transposed (Np x KpP),
// zero-padded. Row stride KpP = Kp + 8 so the LDS-resident copy has dword
// stride == 4 (mod 8)  -> conflict-free 16-lane B-fragment loads.
// ---------------------------------------------------------------------------
__global__ void pad_weights(const float* __restrict__ W, bf16* __restrict__ Wt,
                            int din, int dout, int KpP, int Np) {
    int t = blockIdx.x * blockDim.x + threadIdx.x;
    if (t >= KpP * Np) return;
    int n = t / KpP, k = t % KpP;
    float v = (k < din && n < dout) ? W[(size_t)k * dout + n] : 0.0f;
    Wt[(size_t)n * KpP + k] = (bf16)v;
}

// ---------------------------------------------------------------------------
// Per-batch row max of x_velocity (blocks 0..7) and y_velocity (blocks 8..15)
// ---------------------------------------------------------------------------
__global__ void rowmax_kernel(const float* __restrict__ xv,
                              const float* __restrict__ yv,
                              float* __restrict__ mx) {
    __shared__ float red[256];
    int which = blockIdx.x >> 3;
    int b     = blockIdx.x & 7;
    const float* src = which ? yv : xv;
    float m = -3.4e38f;
    for (int i = threadIdx.x; i < Nc; i += 256)
        m = fmaxf(m, src[(size_t)b * Nc + i]);
    red[threadIdx.x] = m;
    __syncthreads();
    for (int s = 128; s > 0; s >>= 1) {
        if (threadIdx.x < s)
            red[threadIdx.x] = fmaxf(red[threadIdx.x], red[threadIdx.x + s]);
        __syncthreads();
    }
    if (threadIdx.x == 0) mx[which * 8 + b] = red[0];
}

// ---------------------------------------------------------------------------
// Feature builder: for each face row, 49 patch positions x 5 fields -> bf16.
// One thread per (row, slot); slots 0..48 compute, 49..51 zero-pad 245..255.
// ---------------------------------------------------------------------------
__device__ __forceinline__ float mask99(float v) { return (v == -99.0f) ? 0.0f : v; }

__global__ void build_features(const float* __restrict__ xv,
                               const float* __restrict__ yv,
                               const float* __restrict__ p,
                               const int*   __restrict__ faces,
                               const float* __restrict__ mx,
                               bf16* __restrict__ A0) {
    int t = blockIdx.x * blockDim.x + threadIdx.x;
    int r = t >> 6;
    int k = t & 63;
    if (r >= Mrows || k >= 52) return;
    bf16* row = A0 + (size_t)r * D0;
    if (k >= 49) {                       // zero pad columns 245..255
        int start = 245 + (k - 49) * 5;
        int end   = start + 5; if (end > 256) end = 256;
        for (int c = start; c < end; ++c) row[c] = (bf16)0.0f;
        return;
    }
    int b = r >> 14;                     // r / NFc
    int f = r & (NFc - 1);
    int n = faces[f];
    int y = n >> 8;                      // NXc == 256
    int x = n & 255;
    int ky = k / 7, kx = k % 7;
    int yy = y + ky - 3, xx = x + kx - 3;

    float f0 = 0.f, f1 = 0.f, f2 = 0.f, f3 = 0.f, f4 = 0.f;
    if (yy >= 0 && yy < NYc && xx >= 0 && xx < NXc) {
        size_t idx = (size_t)b * Nc + (size_t)yy * NXc + xx;
        float pv = p[idx] - P_ATM;
        // r_x on masked x_velocity: d = m[x+1] (replicate last col), u = m[x-1] (inlet 1.0)
        float mu = xv[idx];
        float m  = mask99(mu);
        float dn = (xx + 1 < NXc) ? mask99(xv[idx + 1]) : m;
        float up = (xx == 0) ? 1.0f : mask99(xv[idx - 1]);
        float rx = (m - up) / (dn - m + EPSf);
        rx = fminf(fmaxf(rx, 0.0f), 2.0f);
        // r_y on masked y_velocity: d = m[y-1] (zero at top), u = m[y+1] (zero at bottom)
        float mv  = yv[idx];
        float mm  = mask99(mv);
        float dup = (yy == 0) ? 0.0f : mask99(yv[idx - NXc]);
        float udn = (yy == NYc - 1) ? 0.0f : mask99(yv[idx + NXc]);
        float ry  = (mm - udn) / (dup - mm + EPSf);
        ry = fminf(fmaxf(ry, 0.0f), 2.0f);
        f0 = pv; f1 = rx; f2 = ry;
        f3 = mu / (mx[b] + EPSf);
        f4 = mv / (mx[8 + b] + EPSf);
    }
    int c0 = k * 5;
    row[c0 + 0] = (bf16)f0;
    row[c0 + 1] = (bf16)f1;
    row[c0 + 2] = (bf16)f2;
    row[c0 + 3] = (bf16)f3;
    row[c0 + 4] = (bf16)f4;
}

// ---------------------------------------------------------------------------
// GEMM + bias + tanh:  C[M x Np] = tanh(A[M x Kp] * Wt^T + bias), bf16 in/out.
// Block: 256 threads (8 waves), tile 256(M) x 32(N); each wave: 32x32 via
// 2x2 v_wmma_f32_16x16x32_bf16, K step 32.
//   - B operand: the block's 32-row weight slice (32 x KpP bf16, <=17KB) is
//     DMA'd to LDS once by the Tensor Data Mover, then reused for all K-steps.
//   - A operand: staged through LDS (stride-40 rows) per K-step.
// ISA fragment layout (16-bit A/B): lane = {idx = lane&15, khalf = lane>>4};
// element e -> k = (e/8)*16 + khalf*8 + e%8  ==> two contiguous 16B LDS loads.
// ---------------------------------------------------------------------------
union Frag { uint4 q[2]; v16bf v; };

__global__ void __launch_bounds__(256) gemm_tanh(
    const bf16* __restrict__ A, const bf16* __restrict__ Wt,
    const float* __restrict__ bias, bf16* __restrict__ C,
    int Kp, int KpP, int Np, int dout) {

    extern __shared__ __align__(16) char smem[];
    bf16* As   = (bf16*)smem;                    // 256 x LDS_STRIDE
    bf16* Wlds = (bf16*)(smem + A_SMEM_BYTES);   // 32 x KpP

    const int tid   = threadIdx.x;
    const int wave  = tid >> 5;
    const int lane  = tid & 31;
    const int lrow  = lane & 15;
    const int khalf = lane >> 4;
    const int bm    = blockIdx.x * 256;
    const int bn    = blockIdx.y * 32;

    // --- TDM: pull this block's 32 weight rows (32 x KpP bf16) into LDS ----
    if (tid == 0) {
        unsigned ldsoff = (unsigned)(unsigned long long)(void*)Wlds; // low 32b = LDS offset
        unsigned long long ga =
            (unsigned long long)(const void*)(Wt + (size_t)bn * KpP);
        v4u g0;
        g0[0] = 1u;                                        // count=1, user desc
        g0[1] = ldsoff;                                    // lds_addr
        g0[2] = (unsigned)(ga & 0xFFFFFFFFu);              // global_addr[31:0]
        g0[3] = (unsigned)((ga >> 32) & 0x1FFFFFFull)      // global_addr[56:32]
              | (2u << 30);                                // type = 2 (image)
        v8i g1;
        g1[0] = (int)(1u << 16);                           // data_size=1 (2B)
        g1[1] = (int)(((unsigned)KpP & 0xFFFFu) << 16);    // tensor_dim0[15:0]
        g1[2] = (int)((((unsigned)KpP >> 16) & 0xFFFFu)    // tensor_dim0[31:16]
              | (32u << 16));                              // tensor_dim1[15:0]=32
        g1[3] = (int)(((unsigned)KpP & 0xFFFFu) << 16);    // tile_dim0 = KpP
        g1[4] = (int)32u;                                  // tile_dim1=32, tile_dim2=0
        g1[5] = (int)(unsigned)KpP;                        // tensor_dim0_stride lo32
        g1[6] = 0;                                         // stride hi, dim1_stride lo
        g1[7] = 0;
        v4i gz4 = {0, 0, 0, 0};
        v8i gz8 = {0, 0, 0, 0, 0, 0, 0, 0};
        // 6-arg toolchain form: (g0, g1, g2, g3, g4, cpol)
        __builtin_amdgcn_tensor_load_to_lds(g0, g1, gz4, gz4, gz8, 0);
        __builtin_amdgcn_s_wait_tensorcnt(0);
    }
    __syncthreads();

    v8f acc[2][2];
    #pragma unroll
    for (int mi = 0; mi < 2; ++mi)
        #pragma unroll
        for (int ni = 0; ni < 2; ++ni)
            #pragma unroll
            for (int e = 0; e < 8; ++e) acc[mi][ni][e] = 0.0f;

    for (int kk = 0; kk < Kp; kk += 32) {
        // Stage A tile: 256 rows x 32 bf16 (1024 x 16B chunks)
        #pragma unroll
        for (int i = 0; i < 4; ++i) {
            int c    = i * 256 + tid;
            int row  = c >> 2;
            int part = c & 3;
            *(uint4*)&As[row * LDS_STRIDE + part * 8] =
                *(const uint4*)(A + (size_t)(bm + row) * Kp + kk + part * 8);
        }
        __syncthreads();

        Frag af[2], bfr[2];
        #pragma unroll
        for (int mi = 0; mi < 2; ++mi) {
            int row = wave * 32 + mi * 16 + lrow;
            af[mi].q[0] = *(const uint4*)&As[row * LDS_STRIDE + khalf * 8];
            af[mi].q[1] = *(const uint4*)&As[row * LDS_STRIDE + 16 + khalf * 8];
        }
        #pragma unroll
        for (int ni = 0; ni < 2; ++ni) {
            int rowb = ni * 16 + lrow;
            bfr[ni].q[0] = *(const uint4*)&Wlds[rowb * KpP + kk + khalf * 8];
            bfr[ni].q[1] = *(const uint4*)&Wlds[rowb * KpP + kk + 16 + khalf * 8];
        }
        #pragma unroll
        for (int mi = 0; mi < 2; ++mi)
            #pragma unroll
            for (int ni = 0; ni < 2; ++ni)
                acc[mi][ni] = __builtin_amdgcn_wmma_f32_16x16x32_bf16(
                    false, af[mi].v, false, bfr[ni].v,
                    (short)0, acc[mi][ni], false, false);
        __syncthreads();
    }

    // Epilogue: bias + tanh + bf16 store.
    // C/D layout: lane -> n = lane&15, element r -> m = r + 8*(lane>>4).
    #pragma unroll
    for (int ni = 0; ni < 2; ++ni) {
        int col = bn + ni * 16 + lrow;
        float bv = (col < dout) ? bias[col] : 0.0f;
        #pragma unroll
        for (int mi = 0; mi < 2; ++mi) {
            int mbase = bm + wave * 32 + mi * 16 + khalf * 8;
            #pragma unroll
            for (int r2 = 0; r2 < 8; ++r2) {
                float v = tanhf(acc[mi][ni][r2] + bv);
                C[(size_t)(mbase + r2) * Np + col] = (bf16)v;
            }
        }
    }
}

// ---------------------------------------------------------------------------
// Heads + scatter. G: [Mrows x 32] bf16 (first 10 valid). x-set: center patch
// row (dy=0, dx=-2..2); y-set: center column (dy=-2..2, dx=0). Taps on raw
// x/y velocity with SAME zero padding. Writes both outputs.
// ---------------------------------------------------------------------------
__global__ void heads_scatter(const bf16* __restrict__ G,
                              const float* __restrict__ xv,
                              const float* __restrict__ yv,
                              const int*   __restrict__ faces,
                              float* __restrict__ out, int isY) {
    int t = blockIdx.x * blockDim.x + threadIdx.x;
    if (t >= Mrows) return;
    int b = t >> 14;
    int f = t & (NFc - 1);
    int n = faces[f];
    int y = n >> 8, x = n & 255;
    const bf16* g = G + (size_t)t * D6;
    float ra = 0.0f, rb = 0.0f;
    if (!isY) {
        #pragma unroll
        for (int k = 0; k < 5; ++k) {
            int xx = x - 2 + k;
            float uv = 0.f, vv = 0.f;
            if (xx >= 0 && xx < NXc) {
                size_t idx = (size_t)b * Nc + (size_t)y * NXc + xx;
                uv = xv[idx]; vv = yv[idx];
            }
            ra += (float)g[k]     * uv;
            rb += (float)g[5 + k] * vv;
        }
    } else {
        #pragma unroll
        for (int k = 0; k < 5; ++k) {
            int yy = y - 2 + k;
            float uv = 0.f, vv = 0.f;
            if (yy >= 0 && yy < NYc) {
                size_t idx = (size_t)b * Nc + (size_t)yy * NXc + x;
                uv = xv[idx]; vv = yv[idx];
            }
            ra += (float)g[k]     * uv;
            rb += (float)g[5 + k] * vv;
        }
    }
    out[(size_t)b * Nc + n]                      = ra;  // x_velocity_edge
    out[(size_t)Bsz * Nc + (size_t)b * Nc + n]   = rb;  // y_velocity_edge
}

// ---------------------------------------------------------------------------
// Host: classify params, carve workspace, launch pipeline.
// ---------------------------------------------------------------------------
extern "C" void kernel_launch(void* const* d_in, const int* in_sizes, int n_in,
                              void* d_out, int out_size, void* d_ws, size_t ws_size,
                              hipStream_t stream) {
    const float* xv = (const float*)d_in[0];
    const float* yv = (const float*)d_in[1];
    const float* pP = (const float*)d_in[2];
    const int* xfaces = (const int*)d_in[3];
    const int* yfaces = (const int*)d_in[4];

    // Resolve pytree flatten order of params from leaf sizes.
    const float *Wl[2][4], *Bl[2][4], *gW[2], *gB[2];
    if (in_sizes[5] == 245 * 214) {
        // insertion order: local_x{W,b}, local_y{W,b}, global{W,b}
        for (int i = 0; i < 4; ++i) {
            Wl[0][i] = (const float*)d_in[5 + i];
            Bl[0][i] = (const float*)d_in[9 + i];
            Wl[1][i] = (const float*)d_in[13 + i];
            Bl[1][i] = (const float*)d_in[17 + i];
        }
        gW[0] = (const float*)d_in[21]; gW[1] = (const float*)d_in[22];
        gB[0] = (const float*)d_in[23]; gB[1] = (const float*)d_in[24];
    } else {
        // alphabetical: global{W,b}, local_x{W,b}, local_y{W,b}
        gW[0] = (const float*)d_in[5]; gW[1] = (const float*)d_in[6];
        gB[0] = (const float*)d_in[7]; gB[1] = (const float*)d_in[8];
        for (int i = 0; i < 4; ++i) {
            Wl[0][i] = (const float*)d_in[9 + i];
            Bl[0][i] = (const float*)d_in[13 + i];
            Wl[1][i] = (const float*)d_in[17 + i];
            Bl[1][i] = (const float*)d_in[21 + i];
        }
    }

    // Workspace carve (256B aligned).
    size_t off = 0;
    auto carve = [&](size_t bytes) -> void* {
        void* ptr = (char*)d_ws + off;
        off = (off + bytes + 255) & ~(size_t)255;
        return ptr;
    };
    const int Kdim[6]  = { D0, D1, D2, D3, D4, D5 };
    const int Ndim[6]  = { D1, D2, D3, D4, D5, D6 };
    const int dinr[6]  = { 245, 214, 185, 156, 127, 68 };
    const int doutr[6] = { 214, 185, 156, 127, 68, 10 };
    int KdimP[6];
    for (int l = 0; l < 6; ++l) KdimP[l] = Kdim[l] + 8;   // padded row stride

    bf16* WtL[2][4];
    for (int s = 0; s < 2; ++s)
        for (int i = 0; i < 4; ++i)
            WtL[s][i] = (bf16*)carve((size_t)Ndim[i] * KdimP[i] * sizeof(bf16));
    bf16* WtG[2];
    WtG[0] = (bf16*)carve((size_t)Ndim[4] * KdimP[4] * sizeof(bf16));
    WtG[1] = (bf16*)carve((size_t)Ndim[5] * KdimP[5] * sizeof(bf16));
    float* mx = (float*)carve(16 * sizeof(float));
    bf16* bufA = (bf16*)carve((size_t)Mrows * D0 * sizeof(bf16));   // 64 MB
    bf16* bufB = (bf16*)carve((size_t)Mrows * D1 * sizeof(bf16));   // 56 MB

    // 1) weight prep (transposed, padded bf16, stride KpP)
    for (int s = 0; s < 2; ++s)
        for (int i = 0; i < 4; ++i) {
            int total = KdimP[i] * Ndim[i];
            pad_weights<<<(total + 255) / 256, 256, 0, stream>>>(
                Wl[s][i], WtL[s][i], dinr[i], doutr[i], KdimP[i], Ndim[i]);
        }
    {
        int t0 = KdimP[4] * Ndim[4];
        pad_weights<<<(t0 + 255) / 256, 256, 0, stream>>>(gW[0], WtG[0], 127, 68, KdimP[4], Ndim[4]);
        int t1 = KdimP[5] * Ndim[5];
        pad_weights<<<(t1 + 255) / 256, 256, 0, stream>>>(gW[1], WtG[1], 68, 10, KdimP[5], Ndim[5]);
    }

    // 2) per-batch maxima
    rowmax_kernel<<<16, 256, 0, stream>>>(xv, yv, mx);

    // 3) per face-set: features -> 6 WMMA GEMMs (TDM weight prefetch) -> heads
    for (int s = 0; s < 2; ++s) {
        const int* faces = s ? yfaces : xfaces;

        build_features<<<(Mrows * 64) / 256, 256, 0, stream>>>(
            xv, yv, pP, faces, mx, bufA);

        const bf16* Win[6]  = { WtL[s][0], WtL[s][1], WtL[s][2], WtL[s][3], WtG[0], WtG[1] };
        const float* bin[6] = { Bl[s][0], Bl[s][1], Bl[s][2], Bl[s][3], gB[0], gB[1] };
        bf16* acts[2] = { bufA, bufB };
        for (int l = 0; l < 6; ++l) {
            const bf16* Ain = acts[l & 1];
            bf16* Aout      = acts[(l + 1) & 1];
            dim3 grid(Mrows / 256, Ndim[l] / 32);
            size_t smembytes = (size_t)A_SMEM_BYTES + (size_t)32 * KdimP[l] * sizeof(bf16);
            gemm_tanh<<<grid, 256, smembytes, stream>>>(
                Ain, Win[l], bin[l], Aout, Kdim[l], KdimP[l], Ndim[l], doutr[l]);
        }
        // after 6 layers output lands in bufA (even index)
        heads_scatter<<<Mrows / 256, 256, 0, stream>>>(
            bufA, xv, yv, faces, (float*)d_out, s);
    }
}
